// NE_AGN_17617955848434
// MI455X (gfx1250) — matrix operand order repaired and checked by reading, not compiled, not run on hardware
//
#include <hip/hip_runtime.h>
#include <math.h>

// Problem constants (match reference)
#define B_TOT   65536
#define M_MOD   8
#define F_DIM   128
#define ALPHA   0.2f
#define NEGBIG  -9.0e15f

// Tiling
#define WAVES      8
#define ITERS      4
#define BATCH_WG   (WAVES * 2)      // 16 batches per WG iteration
#define LDS_STRIDE 136              // padded row stride (elems) for W^T in LDS

typedef __attribute__((ext_vector_type(16))) __bf16 v16bf;
typedef __attribute__((ext_vector_type(8)))  float  v8f;

union AFrag { unsigned u[8]; v16bf v; };

__device__ __forceinline__ unsigned short cvt_bf16(float x) {
  unsigned a = __float_as_uint(x);
  a = (a + 0x7FFFu + ((a >> 16) & 1u)) >> 16;   // round-to-nearest-even
  return (unsigned short)a;
}
// round-half-up to bf16 and pack two floats' high halves with one v_perm_b32
__device__ __forceinline__ unsigned pk_bf16(float lo, float hi) {
  const unsigned a = __float_as_uint(lo) + 0x8000u;
  const unsigned b = __float_as_uint(hi) + 0x8000u;
  return __builtin_amdgcn_perm(b, a, 0x07060302u);  // {b[3],b[2],a[3],a[2]}
}

__global__ __launch_bounds__(256, 4)   // cap VGPRs (<256) -> no vgpr-msb traffic
void gat_fused(const float* __restrict__ h, const float* __restrict__ adj,
               const float* __restrict__ W, const float* __restrict__ a_vec,
               const float* __restrict__ a_node, float* __restrict__ out) {
  // W^T as bf16 (padded rows: bank-conflict-free b128 reads), plus the small
  // constant vectors so no per-lane constants persist in VGPRs.
  __shared__ unsigned short wt[F_DIM * LDS_STRIDE];   // 34 KB
  __shared__ float an_l[F_DIM];                       // a_node
  __shared__ float av_l[2 * F_DIM];                   // a1 | a2

  const int tid = threadIdx.x;
  for (int i = tid; i < F_DIM * F_DIM; i += 256) {
    const int k = i >> 7, n = i & 127;            // W is [k][n] row-major
    wt[n * LDS_STRIDE + k] = cvt_bf16(W[i]);      // store transposed
  }
  if (tid < F_DIM) an_l[tid] = a_node[tid];
  if (tid < 2 * F_DIM) av_l[tid] = a_vec[tid];

  const int wave  = tid >> 5;
  const int lane  = tid & 31;
  const int ln    = lane & 15;       // column lane / A-row
  const int hi    = lane >> 4;       // lane half (0/1)
  const int kbase = hi * 8;          // A-fragment K sub-base per half

  #pragma unroll 1
  for (int it = 0; it < ITERS; ++it) {
    // LDS barrier: orders the one-time fill and keeps all ds_reads (B tile,
    // a_node, a) inside the iteration so nothing gets hoisted into VGPRs.
    __syncthreads();

    const int bblk = (blockIdx.x * ITERS + it) * BATCH_WG;
    const int b0   = bblk + wave * 2;            // two batches per wave

    // ---- stream h rows in A-fragment layout: row = ln, chunks @ 16c+kbase.
    //      Fused: node-attention dot (fp32) + bf16 pack; fp32 h never persists.
    const float* hrow =
        h + ((size_t)(b0 + (ln >> 3)) * M_MOD + (ln & 7)) * F_DIM;
    AFrag afr[4];
    float pe = 0.f;
    #pragma unroll
    for (int c = 0; c < 8; ++c) {
      const float4 x  = *(const float4*)(hrow + 16*c + kbase);
      const float4 y  = *(const float4*)(hrow + 16*c + kbase + 4);
      const float4 ax = *(const float4*)(an_l + 16*c + kbase);
      const float4 ay = *(const float4*)(an_l + 16*c + kbase + 4);
      pe += x.x*ax.x + x.y*ax.y + x.z*ax.z + x.w*ax.w
          + y.x*ay.x + y.y*ay.y + y.z*ay.z + y.w*ay.w;
      const int s = c >> 1, hf = (c & 1) * 4;
      afr[s].u[hf+0] = pk_bf16(x.x, x.y);
      afr[s].u[hf+1] = pk_bf16(x.z, x.w);
      afr[s].u[hf+2] = pk_bf16(y.x, y.y);
      afr[s].u[hf+3] = pk_bf16(y.z, y.w);
    }
    if (it + 1 < ITERS) {                        // global_prefetch_b8
      __builtin_prefetch(hrow + (size_t)BATCH_WG * M_MOD * F_DIM, 0, 1);
    }

    // ---- node attention softmax over the 8 modalities ----
    pe += __shfl_xor(pe, 16, 32);                // lane pair holds same row
    float le = pe > 0.f ? pe : ALPHA * pe;       // leaky relu
    float mx = le;
    mx = fmaxf(mx, __shfl_xor(mx, 1, 32));
    mx = fmaxf(mx, __shfl_xor(mx, 2, 32));
    mx = fmaxf(mx, __shfl_xor(mx, 4, 32));
    const float ex = __expf(le - mx);
    float sm = ex;
    sm += __shfl_xor(sm, 1, 32);
    sm += __shfl_xor(sm, 2, 32);
    sm += __shfl_xor(sm, 4, 32);
    // residual + attention weight (v_rcp_f32 instead of full IEEE divide)
    const float scale = 1.f + ex * __builtin_amdgcn_rcpf(sm);
    // per-modality scales for this lane's batch (C-layout row v = modality v)
    float sc[8];
    #pragma unroll
    for (int v = 0; v < 8; ++v) sc[v] = __shfl(scale, 8 * hi + v, 32);

    // ---- Wh = h x W (16x128 tile: 8 N-tiles x 4 K-steps of 32) ----
    v8f acc[8];
    #pragma unroll
    for (int nt = 0; nt < 8; ++nt) {
      acc[nt] = (v8f){0.f,0.f,0.f,0.f,0.f,0.f,0.f,0.f};
      #pragma unroll
      for (int s = 0; s < 4; ++s) {
        AFrag bfr;
        const uint4* bp = (const uint4*)(
            wt + (nt * 16 + ln) * LDS_STRIDE + s * 32 + hi * 8);
        const uint4 q0 = bp[0], q1 = bp[2];
        bfr.u[0]=q0.x; bfr.u[1]=q0.y; bfr.u[2]=q0.z; bfr.u[3]=q0.w;
        bfr.u[4]=q1.x; bfr.u[5]=q1.y; bfr.u[6]=q1.z; bfr.u[7]=q1.w;
        acc[nt] = __builtin_amdgcn_wmma_f32_16x16x32_bf16(
            false, afr[s].v, false, bfr.v, (short)0, acc[nt], false, false);
      }
    }
    // apply node-attention row scale in f32: diag(s)·(hW) == (diag(s)h)·W
    #pragma unroll
    for (int nt = 0; nt < 8; ++nt) {
      #pragma unroll
      for (int v = 0; v < 8; ++v) acc[nt][v] *= sc[v];
    }

    // ---- Wh1/Wh2: reduce accumulators against a1/a2 across 16 lanes ----
    float a1v[8], a2v[8];
    #pragma unroll
    for (int nt = 0; nt < 8; ++nt) {
      a1v[nt] = av_l[nt * 16 + ln];
      a2v[nt] = av_l[128 + nt * 16 + ln];
    }
    float r1[8], r2[8];
    #pragma unroll
    for (int v = 0; v < 8; ++v) {
      float s1 = 0.f, s2 = 0.f;
      #pragma unroll
      for (int nt = 0; nt < 8; ++nt) {
        s1 += acc[nt][v] * a1v[nt];
        s2 += acc[nt][v] * a2v[nt];
      }
      s1 += __shfl_xor(s1, 1, 32); s1 += __shfl_xor(s1, 2, 32);
      s1 += __shfl_xor(s1, 4, 32); s1 += __shfl_xor(s1, 8, 32);
      s2 += __shfl_xor(s2, 1, 32); s2 += __shfl_xor(s2, 2, 32);
      s2 += __shfl_xor(s2, 4, 32); s2 += __shfl_xor(s2, 8, 32);
      r1[v] = s1; r2[v] = s2;      // Wh1/Wh2 for modality v of this lane's batch
    }

    // ---- masked edge softmax + h' = att x Wh + ReLU, stream out ----
    const int    bb   = b0 + hi;                 // lane-half selects batch
    const float* adjp = adj + (size_t)bb * (M_MOD * M_MOD);
    float*       outp = out + (size_t)bb * (M_MOD * F_DIM);
    #pragma unroll
    for (int i = 0; i < 8; ++i) {
      const float4 m0 = *(const float4*)(adjp + i * 8);
      const float4 m1 = *(const float4*)(adjp + i * 8 + 4);
      const float av[8] = {m0.x,m0.y,m0.z,m0.w,m1.x,m1.y,m1.z,m1.w};
      float ev[8];
      float emax = NEGBIG;
      #pragma unroll
      for (int j = 0; j < 8; ++j) {
        ev[j] = (av[j] > 0.f) ? (r1[j] + r2[i]) : NEGBIG;
        emax = fmaxf(emax, ev[j]);
      }
      float esum = 0.f;
      #pragma unroll
      for (int j = 0; j < 8; ++j) { ev[j] = __expf(ev[j] - emax); esum += ev[j]; }
      const float inv = __builtin_amdgcn_rcpf(esum);   // v_rcp_f32
      #pragma unroll
      for (int nt = 0; nt < 8; ++nt) {
        float hp = 0.f;
        #pragma unroll
        for (int j = 0; j < 8; ++j) hp += ev[j] * acc[nt][j];
        hp *= inv;
        outp[i * F_DIM + nt * 16 + ln] = fmaxf(hp, 0.f);   // ReLU (concat=True)
      }
    }
  }
}

extern "C" void kernel_launch(void* const* d_in, const int* in_sizes, int n_in,
                              void* d_out, int out_size, void* d_ws, size_t ws_size,
                              hipStream_t stream) {
  (void)in_sizes; (void)n_in; (void)out_size; (void)d_ws; (void)ws_size;
  const float* h      = (const float*)d_in[0];   // [B, M, F]
  const float* adj    = (const float*)d_in[1];   // [B, M, M]
  const float* W      = (const float*)d_in[2];   // [F, F]
  const float* a_vec  = (const float*)d_in[3];   // [2F, 1]
  const float* a_node = (const float*)d_in[4];   // [F, 1]
  float* out = (float*)d_out;                    // [B, M, F]

  const int grid = B_TOT / (ITERS * BATCH_WG);   // 1024 workgroups
  gat_fused<<<grid, 256, 0, stream>>>(h, adj, W, a_vec, a_node, out);
}
